// LlamaAttention_heavy_hitter_7584912245150
// MI455X (gfx1250) — matrix operand order repaired and checked by reading, not compiled, not run on hardware
//
#include <hip/hip_runtime.h>
#include <math.h>

typedef __bf16 bf16_t;
typedef __attribute__((ext_vector_type(16))) __bf16 v16bf;
typedef __attribute__((ext_vector_type(8)))  __bf16 v8bf;
typedef __attribute__((ext_vector_type(8)))  float  v8f;
typedef unsigned int u32x4 __attribute__((ext_vector_type(4)));
typedef int          i32x4 __attribute__((ext_vector_type(4)));
typedef int          i32x8 __attribute__((ext_vector_type(8)));

#define SEQ    2048
#define HID    4096
#define NHEAD  32
#define HDIM   128
#define QTILES (SEQ / 16)   // 128
#define HEAVY  204
#define RECENT 204
#define QKSCALE 0.08838834764831845f  // 1/sqrt(128)
#define BIGNEG  (-3.0e38f)

#if __has_builtin(__builtin_amdgcn_tensor_load_to_lds)
#define USE_TDM 1
#else
#define USE_TDM 0
#endif

// ---------------------------------------------------------------------------
// WMMA helpers (CDNA5 16x16x32 bf16, f32 accumulate)
// ---------------------------------------------------------------------------
static __device__ __forceinline__ v8f wmma_bf16(v16bf a, v16bf b, v8f c) {
  return __builtin_amdgcn_wmma_f32_16x16x32_bf16(false, a, false, b, (short)0, c,
                                                 false, false);
}

// A-fragment (16x32 bf16): lane holds row m=(lane&15), one contiguous
// 16-element K chunk at K = 16*hi (hi = lane>>4).
static __device__ __forceinline__ v16bf load_a_frag(const bf16_t* row, int hi) {
  return *(const v16bf*)(row + 16 * hi);
}

// ---------------------------------------------------------------------------
// Tensor Data Mover: 2D tile (tile_w x tile_h elems, 2B each) -> LDS.
// D# packing per CDNA5 ISA; 6-arg builtin (clang-23).
// ---------------------------------------------------------------------------
static __device__ __forceinline__ void tdm_load_tile_2d(
    const void* tile_base, unsigned lds_off, unsigned tensor_w,
    unsigned tensor_h, unsigned tile_w, unsigned tile_h,
    unsigned row_stride_elems) {
#if USE_TDM
  unsigned long long ga = (unsigned long long)tile_base;
  u32x4 g0 = {1u,                                   // count=1 (valid), user mode
              lds_off,                              // lds_addr (bytes)
              (unsigned)(ga & 0xFFFFFFFFull),       // global_addr[31:0]
              (unsigned)((ga >> 32) & 0x01FFFFFFull) | 0x80000000u};  // type=2
  i32x8 g1 = {(int)(1u << 16),                      // data_size = 2 bytes
              (int)((tensor_w & 0xFFFFu) << 16),    // tensor_dim0[15:0]
              (int)(((tensor_w >> 16) & 0xFFFFu) |  // tensor_dim0[31:16]
                    ((tensor_h & 0xFFFFu) << 16)),  // tensor_dim1[15:0]
              (int)(((tensor_h >> 16) & 0xFFFFu) |  // tensor_dim1[31:16]
                    ((tile_w & 0xFFFFu) << 16)),    // tile_dim0
              (int)(tile_h & 0xFFFFu),              // tile_dim1 (tile_dim2=0)
              (int)row_stride_elems,                // tensor_dim0_stride[31:0]
              0, 0};
  i32x4 gz4 = {0, 0, 0, 0};
  i32x8 gz8 = {0, 0, 0, 0, 0, 0, 0, 0};
  __builtin_amdgcn_tensor_load_to_lds(g0, g1, gz4, gz4, gz8, 0);
#else
  (void)tile_base; (void)lds_off; (void)tensor_w; (void)tensor_h;
  (void)tile_w; (void)tile_h; (void)row_stride_elems;
#endif
}

// ---------------------------------------------------------------------------
// Elementwise f32 -> bf16 convert
// ---------------------------------------------------------------------------
__global__ void cvt_bf16_kernel(const float* __restrict__ in,
                                bf16_t* __restrict__ out, int n) {
  int i = blockIdx.x * blockDim.x + threadIdx.x;
  if (i < n) out[i] = (bf16_t)in[i];
}

// Transpose-convert: out[c][r] = (bf16) in[r][c], square HID x HID.
__global__ __launch_bounds__(256) void transpose_bf16_kernel(
    const float* __restrict__ in, bf16_t* __restrict__ out) {
  __shared__ float tile[32][33];
  const int bx = blockIdx.x * 32, by = blockIdx.y * 32;
  const int tx = threadIdx.x, ty = threadIdx.y;  // 32 x 8
#pragma unroll
  for (int i = 0; i < 32; i += 8)
    tile[ty + i][tx] = in[(size_t)(by + ty + i) * HID + bx + tx];
  __syncthreads();
#pragma unroll
  for (int i = 0; i < 32; i += 8)
    out[(size_t)(bx + ty + i) * HID + by + tx] = (bf16_t)tile[tx][ty + i];
}

// ---------------------------------------------------------------------------
// GEMM: C[M][N] = A[M][K] * Bt[N][K]^T  (bf16 in, f32 acc, OutT out).
// 256 threads = 8 waves (4 row x 2 col), block tile 128x64, k-step 64.
// Double-buffered TDM pipeline: DMA of stage s+1 overlaps WMMA of stage s;
// one s_wait_tensorcnt + one barrier per stage.
// ---------------------------------------------------------------------------
template <typename OutT>
__global__ __launch_bounds__(256) void gemm_bf16_wmma(
    const bf16_t* __restrict__ A, const bf16_t* __restrict__ Bt,
    OutT* __restrict__ C, int M, int N, int K) {
  const int lane = threadIdx.x & 31;
  const int wave = threadIdx.x >> 5;
  const int lm = lane & 15, lh = lane >> 4;
  const int wr = wave & 3, wc = wave >> 2;
  const int rowB = blockIdx.y * 128;
  const int colB = blockIdx.x * 64;
  const int row0 = rowB + wr * 32;
  const int col0 = colB + wc * 32;

  v8f c00 = {0, 0, 0, 0, 0, 0, 0, 0};
  v8f c01 = c00, c10 = c00, c11 = c00;

#if USE_TDM
  __shared__ __align__(64) bf16_t As[2][128 * 64];  // 2 x 16 KB
  __shared__ __align__(64) bf16_t Bs[2][64 * 64];   // 2 x  8 KB
  const unsigned asOff[2] = {
      (unsigned)(unsigned long long)(const void*)&As[0][0],
      (unsigned)(unsigned long long)(const void*)&As[1][0]};
  const unsigned bsOff[2] = {
      (unsigned)(unsigned long long)(const void*)&Bs[0][0],
      (unsigned)(unsigned long long)(const void*)&Bs[1][0]};

  // prologue: stage 0
  if (wave == 0) {
    tdm_load_tile_2d(A + (size_t)rowB * K, asOff[0], K, M, 64, 128, K);
    tdm_load_tile_2d(Bt + (size_t)colB * K, bsOff[0], K, N, 64, 64, K);
    __builtin_amdgcn_s_wait_tensorcnt(0);
  }
  __syncthreads();

  int buf = 0;
  for (int k = 0; k < K; k += 64) {
    if (wave == 0 && k + 64 < K) {  // prefetch next stage into other buffer
      tdm_load_tile_2d(A + (size_t)rowB * K + k + 64, asOff[buf ^ 1], K, M, 64,
                       128, K);
      tdm_load_tile_2d(Bt + (size_t)colB * K + k + 64, bsOff[buf ^ 1], K, N, 64,
                       64, K);
    }
    const bf16_t* Ab = &As[buf][0];
    const bf16_t* Bb = &Bs[buf][0];
    const bf16_t* a0p = Ab + (size_t)(wr * 32 + lm) * 64;
    const bf16_t* a1p = Ab + (size_t)(wr * 32 + 16 + lm) * 64;
    const bf16_t* b0p = Bb + (size_t)(wc * 32 + lm) * 64;
    const bf16_t* b1p = Bb + (size_t)(wc * 32 + 16 + lm) * 64;
#pragma unroll
    for (int kk = 0; kk < 64; kk += 32) {
      v16bf a0 = load_a_frag(a0p + kk, lh);
      v16bf a1 = load_a_frag(a1p + kk, lh);
      v16bf b0 = *(const v16bf*)(b0p + kk + 16 * lh);
      v16bf b1 = *(const v16bf*)(b1p + kk + 16 * lh);
      c00 = wmma_bf16(a0, b0, c00);
      c01 = wmma_bf16(a0, b1, c01);
      c10 = wmma_bf16(a1, b0, c10);
      c11 = wmma_bf16(a1, b1, c11);
    }
    if (wave == 0) __builtin_amdgcn_s_wait_tensorcnt(0);
    __syncthreads();
    buf ^= 1;
  }
#else
  const bf16_t* pa0 = A + (size_t)(row0 + lm) * K;
  const bf16_t* pa1 = A + (size_t)(row0 + 16 + lm) * K;
  const bf16_t* pb0 = Bt + (size_t)(col0 + lm) * K;
  const bf16_t* pb1 = Bt + (size_t)(col0 + 16 + lm) * K;
  for (int k = 0; k < K; k += 32) {
    __builtin_prefetch(pa0 + k + 256, 0, 1);
    __builtin_prefetch(pb0 + k + 256, 0, 1);
    v16bf a0 = load_a_frag(pa0 + k, lh);
    v16bf a1 = load_a_frag(pa1 + k, lh);
    v16bf b0 = *(const v16bf*)(pb0 + k + 16 * lh);
    v16bf b1 = *(const v16bf*)(pb1 + k + 16 * lh);
    c00 = wmma_bf16(a0, b0, c00);
    c01 = wmma_bf16(a0, b1, c01);
    c10 = wmma_bf16(a1, b0, c10);
    c11 = wmma_bf16(a1, b1, c11);
  }
#endif

#pragma unroll
  for (int r = 0; r < 8; ++r) {
    const int rr0 = row0 + r + 8 * lh;
    const int rr1 = rr0 + 16;
    C[(size_t)rr0 * N + col0 + lm]      = (OutT)c00[r];
    C[(size_t)rr0 * N + col0 + 16 + lm] = (OutT)c01[r];
    C[(size_t)rr1 * N + col0 + lm]      = (OutT)c10[r];
    C[(size_t)rr1 * N + col0 + 16 + lm] = (OutT)c11[r];
  }
}

// ---------------------------------------------------------------------------
// RoPE + layout shuffle: Qb/Kb [h][s][d] bf16 (rotated), Vt [h][d][s] bf16
// ---------------------------------------------------------------------------
__global__ __launch_bounds__(64) void rope_kernel(
    const bf16_t* __restrict__ qin, const bf16_t* __restrict__ kin,
    const bf16_t* __restrict__ vin, const int* __restrict__ pos,
    bf16_t* __restrict__ Qb, bf16_t* __restrict__ Kb, bf16_t* __restrict__ Vt) {
  const int s = blockIdx.x;
  const int h = blockIdx.y;
  const int i = threadIdx.x;  // 0..63 (pair index)
  const float p = (float)pos[s];
  const float invf = powf(10000.0f, -(float)(2 * i) / (float)HDIM);
  const float ang = p * invf;
  const float cs = cosf(ang), sn = sinf(ang);

  const size_t ib = (size_t)s * HID + (size_t)h * HDIM;
  const float q0 = (float)qin[ib + i], q1 = (float)qin[ib + i + 64];
  const float k0 = (float)kin[ib + i], k1 = (float)kin[ib + i + 64];

  const size_t ob = ((size_t)h * SEQ + s) * HDIM;
  Qb[ob + i]      = (bf16_t)(q0 * cs - q1 * sn);
  Qb[ob + i + 64] = (bf16_t)(q1 * cs + q0 * sn);
  Kb[ob + i]      = (bf16_t)(k0 * cs - k1 * sn);
  Kb[ob + i + 64] = (bf16_t)(k1 * cs + k0 * sn);

  Vt[((size_t)h * HDIM + i) * SEQ + s]      = vin[ib + i];
  Vt[((size_t)h * HDIM + i + 64) * SEQ + s] = vin[ib + i + 64];
}

// ---------------------------------------------------------------------------
// Attention pass 1: two key tiles per iteration, two independent WMMA chains
// (loads of one chain overlap WMMAs of the other); lane-local online softmax.
// ---------------------------------------------------------------------------
__global__ __launch_bounds__(32) void attn_pass1(
    const bf16_t* __restrict__ Qb, const bf16_t* __restrict__ Kb,
    float* __restrict__ Mrow, float* __restrict__ Lrow) {
  const int t = blockIdx.x;
  const int h = blockIdx.y;
  const int lane = threadIdx.x & 31;
  const int lm = lane & 15, lh = lane >> 4;
  const int q0 = t * 16;

  const bf16_t* qrow = Qb + ((size_t)h * SEQ + q0 + lm) * HDIM;
  v16bf aq[4];
#pragma unroll
  for (int st = 0; st < 4; ++st) aq[st] = load_a_frag(qrow + st * 32, lh);

  float m_loc[8], l_loc[8];
#pragma unroll
  for (int r = 0; r < 8; ++r) { m_loc[r] = BIGNEG; l_loc[r] = 0.0f; }

  auto update_tile = [&](const v8f& sacc, int kbase) {
#pragma unroll
    for (int r = 0; r < 8; ++r) {
      float s = sacc[r] * QKSCALE;
      const int qpos = q0 + r + 8 * lh;
      if (kbase + lm > qpos) s = BIGNEG;
      const float mn = fmaxf(m_loc[r], s);
      l_loc[r] = l_loc[r] * __expf(m_loc[r] - mn) + __expf(s - mn);
      m_loc[r] = mn;
    }
  };

  const v8f zero = {0, 0, 0, 0, 0, 0, 0, 0};
  const int ntiles = t + 1;
  int j = 0;
  for (; j + 2 <= ntiles; j += 2) {
    const int k0 = j * 16;
    const bf16_t* kr0 = Kb + ((size_t)h * SEQ + k0 + lm) * HDIM;
    const bf16_t* kr1 = kr0 + (size_t)16 * HDIM;
    v16bf b00 = *(const v16bf*)(kr0 + 16 * lh);
    v16bf b01 = *(const v16bf*)(kr0 + 32 + 16 * lh);
    v16bf b02 = *(const v16bf*)(kr0 + 64 + 16 * lh);
    v16bf b03 = *(const v16bf*)(kr0 + 96 + 16 * lh);
    v16bf b10 = *(const v16bf*)(kr1 + 16 * lh);
    v16bf b11 = *(const v16bf*)(kr1 + 32 + 16 * lh);
    v16bf b12 = *(const v16bf*)(kr1 + 64 + 16 * lh);
    v16bf b13 = *(const v16bf*)(kr1 + 96 + 16 * lh);
    v8f s0 = zero, s1 = zero;
    s0 = wmma_bf16(aq[0], b00, s0);
    s1 = wmma_bf16(aq[0], b10, s1);
    s0 = wmma_bf16(aq[1], b01, s0);
    s1 = wmma_bf16(aq[1], b11, s1);
    s0 = wmma_bf16(aq[2], b02, s0);
    s1 = wmma_bf16(aq[2], b12, s1);
    s0 = wmma_bf16(aq[3], b03, s0);
    s1 = wmma_bf16(aq[3], b13, s1);
    update_tile(s0, k0);
    update_tile(s1, k0 + 16);
  }
  if (j < ntiles) {  // tail tile
    const int k0 = j * 16;
    const bf16_t* kr0 = Kb + ((size_t)h * SEQ + k0 + lm) * HDIM;
    v16bf b00 = *(const v16bf*)(kr0 + 16 * lh);
    v16bf b01 = *(const v16bf*)(kr0 + 32 + 16 * lh);
    v16bf b02 = *(const v16bf*)(kr0 + 64 + 16 * lh);
    v16bf b03 = *(const v16bf*)(kr0 + 96 + 16 * lh);
    v8f s0 = zero;
    s0 = wmma_bf16(aq[0], b00, s0);
    s0 = wmma_bf16(aq[1], b01, s0);
    s0 = wmma_bf16(aq[2], b02, s0);
    s0 = wmma_bf16(aq[3], b03, s0);
    update_tile(s0, k0);
  }

#pragma unroll
  for (int r = 0; r < 8; ++r) {
    float m = m_loc[r];
    m = fmaxf(m, __shfl_xor(m, 1));
    m = fmaxf(m, __shfl_xor(m, 2));
    m = fmaxf(m, __shfl_xor(m, 4));
    m = fmaxf(m, __shfl_xor(m, 8));
    float l = l_loc[r] * __expf(m_loc[r] - m);  // 0 for fully-masked lanes
    l += __shfl_xor(l, 1);
    l += __shfl_xor(l, 2);
    l += __shfl_xor(l, 4);
    l += __shfl_xor(l, 8);
    if (lm == 0) {
      const size_t idx = (size_t)h * SEQ + q0 + r + 8 * lh;
      Mrow[idx] = m;
      Lrow[idx] = l;
    }
  }
}

// ---------------------------------------------------------------------------
// Attention pass 2: probs -> (column-sum partials, P @ V via WMMA) -> attnb.
// Two interleaved S chains per 32-key pair.
// ---------------------------------------------------------------------------
__global__ __launch_bounds__(32) void attn_pass2(
    const bf16_t* __restrict__ Qb, const bf16_t* __restrict__ Kb,
    const bf16_t* __restrict__ Vt, const float* __restrict__ Mrow,
    const float* __restrict__ Lrow, float* __restrict__ partial,
    bf16_t* __restrict__ attnb) {
  const int t = blockIdx.x;
  const int h = blockIdx.y;
  const int lane = threadIdx.x & 31;
  const int lm = lane & 15, lh = lane >> 4;
  const int q0 = t * 16;

  const bf16_t* qrow = Qb + ((size_t)h * SEQ + q0 + lm) * HDIM;
  v16bf aq[4];
#pragma unroll
  for (int st = 0; st < 4; ++st) aq[st] = load_a_frag(qrow + st * 32, lh);

  float mr[8], li[8];
#pragma unroll
  for (int r = 0; r < 8; ++r) {
    const size_t idx = (size_t)h * SEQ + q0 + r + 8 * lh;
    mr[r] = Mrow[idx];
    li[r] = 1.0f / Lrow[idx];
  }

  const v8f zero = {0, 0, 0, 0, 0, 0, 0, 0};
  v8f oacc[8];
#pragma unroll
  for (int nb = 0; nb < 8; ++nb) oacc[nb] = zero;

  __shared__ __align__(32) bf16_t pT[16 * 32];  // 16 rows x 32 keys, bf16

  auto emit_probs = [&](const v8f& sacc, int k0v, int half) {
    float colsum = 0.0f;
#pragma unroll
    for (int r = 0; r < 8; ++r) {
      float s = sacc[r] * QKSCALE;
      const int qpos = q0 + r + 8 * lh;
      if (k0v + lm > qpos) s = BIGNEG;
      const float p = __expf(s - mr[r]) * li[r];
      colsum += p;
      pT[(r + 8 * lh) * 32 + half * 16 + lm] = (bf16_t)p;
    }
    colsum += __shfl_xor(colsum, 16);
    if (lh == 0)
      partial[((size_t)h * QTILES + t) * SEQ + k0v + lm] = colsum;
  };

  const int npairs = (t + 2) >> 1;
  for (int jp = 0; jp < npairs; ++jp) {
    const int k0 = jp * 32;
    const bf16_t* kr0 = Kb + ((size_t)h * SEQ + k0 + lm) * HDIM;
    const bf16_t* kr1 = kr0 + (size_t)16 * HDIM;
    v16bf b00 = *(const v16bf*)(kr0 + 16 * lh);
    v16bf b01 = *(const v16bf*)(kr0 + 32 + 16 * lh);
    v16bf b02 = *(const v16bf*)(kr0 + 64 + 16 * lh);
    v16bf b03 = *(const v16bf*)(kr0 + 96 + 16 * lh);
    v16bf b10 = *(const v16bf*)(kr1 + 16 * lh);
    v16bf b11 = *(const v16bf*)(kr1 + 32 + 16 * lh);
    v16bf b12 = *(const v16bf*)(kr1 + 64 + 16 * lh);
    v16bf b13 = *(const v16bf*)(kr1 + 96 + 16 * lh);
    v8f s0 = zero, s1 = zero;
    s0 = wmma_bf16(aq[0], b00, s0);
    s1 = wmma_bf16(aq[0], b10, s1);
    s0 = wmma_bf16(aq[1], b01, s0);
    s1 = wmma_bf16(aq[1], b11, s1);
    s0 = wmma_bf16(aq[2], b02, s0);
    s1 = wmma_bf16(aq[2], b12, s1);
    s0 = wmma_bf16(aq[3], b03, s0);
    s1 = wmma_bf16(aq[3], b13, s1);
    emit_probs(s0, k0, 0);
    emit_probs(s1, k0 + 16, 1);

    // P (16x32) as A fragment from LDS, V^T rows as B fragments
    v16bf pa = load_a_frag(&pT[lm * 32], lh);
#pragma unroll
    for (int nb = 0; nb < 8; ++nb) {
      const bf16_t* vrow =
          Vt + ((size_t)h * HDIM + nb * 16 + lm) * SEQ + k0 + 16 * lh;
      v16bf bv = *(const v16bf*)vrow;
      oacc[nb] = wmma_bf16(pa, bv, oacc[nb]);
    }
  }
#pragma unroll
  for (int nb = 0; nb < 8; ++nb) {
#pragma unroll
    for (int r = 0; r < 8; ++r) {
      attnb[(size_t)(q0 + r + 8 * lh) * HID + h * HDIM + nb * 16 + lm] =
          (bf16_t)oacc[nb][r];
    }
  }
}

// ---------------------------------------------------------------------------
// previous_scores[h][k] = sum over contributing qtiles of partial[h][t][k]
// ---------------------------------------------------------------------------
__global__ void reduce_scores(const float* __restrict__ partial,
                              float* __restrict__ prev) {
  const int idx = blockIdx.x * blockDim.x + threadIdx.x;
  if (idx >= NHEAD * SEQ) return;
  const int h = idx / SEQ;
  const int k = idx % SEQ;
  float s = 0.0f;
  for (int t = k >> 4; t < QTILES; ++t)
    s += partial[((size_t)h * QTILES + t) * SEQ + k];
  prev[idx] = s;
}

// ---------------------------------------------------------------------------
// preserved mask: keep top HEAVY of first (SEQ-RECENT) scores + recent window.
// ---------------------------------------------------------------------------
__global__ __launch_bounds__(256) void select_heavy(
    const float* __restrict__ prev, float* __restrict__ pres) {
  const int h = blockIdx.x;
  const float* sc = prev + (size_t)h * SEQ;
  float* pv = pres + (size_t)h * SEQ;
  const int ncand = SEQ - RECENT;  // 1844
  const int keep = HEAVY;          // 204

  __shared__ unsigned s_lo, s_hi, s_cnt;
  if (threadIdx.x == 0) { s_lo = 0u; s_hi = 0x7F800000u; }
  __syncthreads();

  for (int it = 0; it < 34; ++it) {
    const unsigned lo = s_lo, hi = s_hi;
    if (lo >= hi) break;
    const unsigned mid = lo + ((hi - lo + 1u) >> 1);
    if (threadIdx.x == 0) s_cnt = 0u;
    __syncthreads();
    unsigned c = 0;
    for (int i = threadIdx.x; i < ncand; i += 256)
      if (__float_as_uint(sc[i]) >= mid) ++c;
    atomicAdd(&s_cnt, c);
    __syncthreads();
    if (threadIdx.x == 0) {
      if (s_cnt >= (unsigned)keep) s_lo = mid; else s_hi = mid - 1u;
    }
    __syncthreads();
  }
  const unsigned V = s_lo;

  if (threadIdx.x == 0) s_cnt = 0u;
  __syncthreads();
  {
    unsigned c = 0;
    for (int i = threadIdx.x; i < ncand; i += 256)
      if (__float_as_uint(sc[i]) > V) ++c;
    atomicAdd(&s_cnt, c);
  }
  __syncthreads();

  for (int i = threadIdx.x; i < ncand; i += 256)
    pv[i] = (__float_as_uint(sc[i]) > V) ? 1.0f : 0.0f;
  for (int i = ncand + threadIdx.x; i < SEQ; i += 256) pv[i] = 1.0f;
  __syncthreads();

  if (threadIdx.x == 0) {
    int rem = keep - (int)s_cnt;
    for (int i = ncand - 1; i >= 0 && rem > 0; --i) {
      if (__float_as_uint(sc[i]) == V) { pv[i] = 1.0f; --rem; }
    }
  }
}

// ---------------------------------------------------------------------------
// Host launch
// ---------------------------------------------------------------------------
extern "C" void kernel_launch(void* const* d_in, const int* in_sizes, int n_in,
                              void* d_out, int out_size, void* d_ws,
                              size_t ws_size, hipStream_t stream) {
  (void)in_sizes; (void)n_in; (void)out_size; (void)ws_size;
  const float* hs  = (const float*)d_in[0];
  const int*   pos = (const int*)d_in[1];
  const float* Wq  = (const float*)d_in[2];
  const float* Wk  = (const float*)d_in[3];
  const float* Wv  = (const float*)d_in[4];
  const float* Wo  = (const float*)d_in[5];

  char* ws = (char*)d_ws;
  const size_t SZ_ACT = (size_t)SEQ * HID * sizeof(bf16_t);   // 16 MB
  const size_t SZ_W   = (size_t)HID * HID * sizeof(bf16_t);   // 32 MB

  bf16_t* hsb  = (bf16_t*)(ws);                     // [s][4096] bf16
  bf16_t* WT0  = (bf16_t*)(ws + SZ_ACT);            // Wq^T, later Wo^T
  bf16_t* WT1  = (bf16_t*)(ws + SZ_ACT + SZ_W);     // Wk^T
  bf16_t* WT2  = (bf16_t*)(ws + SZ_ACT + 2 * SZ_W); // Wv^T
  bf16_t* Qpre = (bf16_t*)(ws + SZ_ACT + 3 * SZ_W); // pre-RoPE Q
  bf16_t* Kpre = Qpre + (size_t)SEQ * HID;
  bf16_t* Vpre = Kpre + (size_t)SEQ * HID;
  bf16_t* Qb   = hsb;                               // alias: hs consumed
  bf16_t* Kb   = (bf16_t*)(ws + SZ_ACT + 3 * SZ_W + 3 * SZ_ACT);
  bf16_t* Vt   = Kb + (size_t)SEQ * HID;
  float*  Mrow = (float*)((char*)Vt + SZ_ACT);
  float*  Lrow = Mrow + (size_t)NHEAD * SEQ;
  float*  partial = Lrow + (size_t)NHEAD * SEQ;     // [h][qtile][key] 32 MB
  bf16_t* attnb = Qpre;                             // alias: Qpre consumed

  float* out  = (float*)d_out;
  float* prev = out + (size_t)SEQ * HID;
  float* pres = prev + (size_t)NHEAD * SEQ;

  cvt_bf16_kernel<<<(SEQ * HID + 255) / 256, 256, 0, stream>>>(hs, hsb, SEQ * HID);
  dim3 tb(32, 8), tg(HID / 32, HID / 32);
  transpose_bf16_kernel<<<tg, tb, 0, stream>>>(Wq, WT0);
  transpose_bf16_kernel<<<tg, tb, 0, stream>>>(Wk, WT1);
  transpose_bf16_kernel<<<tg, tb, 0, stream>>>(Wv, WT2);

  dim3 gg(HID / 64, SEQ / 128);
  gemm_bf16_wmma<bf16_t><<<gg, 256, 0, stream>>>(hsb, WT0, Qpre, SEQ, HID, HID);
  gemm_bf16_wmma<bf16_t><<<gg, 256, 0, stream>>>(hsb, WT1, Kpre, SEQ, HID, HID);
  gemm_bf16_wmma<bf16_t><<<gg, 256, 0, stream>>>(hsb, WT2, Vpre, SEQ, HID, HID);

  transpose_bf16_kernel<<<tg, tb, 0, stream>>>(Wo, WT0);

  rope_kernel<<<dim3(SEQ, NHEAD), 64, 0, stream>>>(Qpre, Kpre, Vpre, pos, Qb, Kb, Vt);

  attn_pass1<<<dim3(QTILES, NHEAD), 32, 0, stream>>>(Qb, Kb, Mrow, Lrow);
  attn_pass2<<<dim3(QTILES, NHEAD), 32, 0, stream>>>(Qb, Kb, Vt, Mrow, Lrow,
                                                     partial, attnb);

  reduce_scores<<<(NHEAD * SEQ + 255) / 256, 256, 0, stream>>>(partial, prev);

  gemm_bf16_wmma<float><<<gg, 256, 0, stream>>>(attnb, WT0, out, SEQ, HID, HID);

  select_heavy<<<NHEAD, 256, 0, stream>>>(prev, pres);
}
